// AttModule_89258010345804
// MI455X (gfx1250) — matrix operand order repaired and checked by reading, not compiled, not run on hardware
//
#include <hip/hip_runtime.h>

typedef __attribute__((ext_vector_type(16))) __bf16 v16bf;
typedef __attribute__((ext_vector_type(8)))  float  v8f;

constexpr int Cch = 256;
constexpr int Lln = 131072;
constexpr int TS  = 40;    // padded K-stride (bf16) for 32-wide K tiles (80B, 16B-aligned)
constexpr int TS2 = 136;   // padded K-stride (bf16) for 128-wide K tiles (272B, 16B-aligned)

// ---- async global->LDS (CDNA5, ASYNCcnt) with sync fallback -----------------
#if defined(__gfx1250__) && __has_builtin(__builtin_amdgcn_global_load_async_to_lds_b128)
#define HAS_ASYNC_LDS 1
#else
#define HAS_ASYNC_LDS 0
#endif

#if HAS_ASYNC_LDS
typedef int v4i __attribute__((vector_size(16)));
typedef __attribute__((address_space(1))) v4i as1_v4i;   // global
typedef __attribute__((address_space(3))) v4i as3_v4i;   // LDS
#endif

__device__ __forceinline__ void copy16_g2l(void* l, const void* g) {
#if HAS_ASYNC_LDS
  __builtin_amdgcn_global_load_async_to_lds_b128((as1_v4i*)g, (as3_v4i*)l, 0, 0);
#else
  *reinterpret_cast<uint4*>(l) = *reinterpret_cast<const uint4*>(g);
#endif
}
__device__ __forceinline__ void lds_copy_fence() {
#if HAS_ASYNC_LDS
#if __has_builtin(__builtin_amdgcn_s_wait_asynccnt)
  __builtin_amdgcn_s_wait_asynccnt(0);
#else
  asm volatile("s_wait_asynccnt 0x0" ::: "memory");
#endif
#endif
}

// ---- WMMA fragment helpers (CDNA5 ISA 7.12.2, wave32) -----------------------
// A (16-bit, 16x32): lane m=lane%16, half=lane/16; elems 0..7 <- K=8h+0..7, 8..15 <- K=16+8h+0..7
__device__ __forceinline__ v16bf frag_a(const __bf16* rowk, int half) {
  v16bf r;
  uint4* d = reinterpret_cast<uint4*>(&r);
  d[0] = *reinterpret_cast<const uint4*>(rowk + 8 * half);
  d[1] = *reinterpret_cast<const uint4*>(rowk + 16 + 8 * half);
  return r;
}
// B (16-bit, 32x16): lane n=lane%16 is column; elems 0..7 <- K=8h+0..7, 8..15 <- K=16+8h+0..7
__device__ __forceinline__ v16bf frag_b(const __bf16* rowk, int half) {
  v16bf r;
  uint4* d = reinterpret_cast<uint4*>(&r);
  d[0] = *reinterpret_cast<const uint4*>(rowk + 8 * half);
  d[1] = *reinterpret_cast<const uint4*>(rowk + 16 + 8 * half);
  return r;
}
__device__ __forceinline__ v8f wmma_bf(v16bf a, v16bf b, v8f c) {
  return __builtin_amdgcn_wmma_f32_16x16x32_bf16(false, a, false, b, (short)0, c, false, false);
}

// ---- K0: weight prep (f32 -> bf16, Wff re-layout [t][o][c]) -----------------
__global__ __launch_bounds__(256) void k_prep(const float* __restrict__ Wff,
    const float* __restrict__ Wv, const float* __restrict__ Wo, const float* __restrict__ Wc,
    __bf16* __restrict__ Wffb, __bf16* __restrict__ Wvb, __bf16* __restrict__ Wob,
    __bf16* __restrict__ Wcb) {
  int i = blockIdx.x * 256 + threadIdx.x;
  if (i < 196608) {               // 3*256*256, Wff input layout [o][c][t]
    int t = i >> 16, r = i & 65535, o = r >> 8, c = r & 255;
    Wffb[i] = (__bf16)Wff[o * 768 + c * 3 + t];
  } else if (i < 229376) { int j = i - 196608; Wvb[j] = (__bf16)Wv[j]; }
  else if (i < 262144)   { int j = i - 229376; Wob[j] = (__bf16)Wo[j]; }
  else if (i < 327680)   { int j = i - 262144; Wcb[j] = (__bf16)Wc[j]; }
}

// ---- K1: dilated conv as 3 chained GEMMs + bias + relu + partial stats ------
// block tile: 64 oc x 64 l, 8 waves, each wave = 1 A-frag x 2 B-frags
__global__ __launch_bounds__(256) void k_conv(const float* __restrict__ x,
    const __bf16* __restrict__ Wffb, const float* __restrict__ bff,
    float* __restrict__ outf, float* __restrict__ psum, float* __restrict__ psq) {
  __shared__ __bf16 A[64 * TS];
  __shared__ __bf16 B[64 * TS];
  int tid  = threadIdx.x;
  int lb   = blockIdx.x * 64;       // L base
  int ob   = blockIdx.y * 64;       // out-channel base
  int wave = tid >> 5, lane = tid & 31;
  int half = lane >> 4, n = lane & 15;
  int wo = wave & 3, wl = wave >> 2;          // oc tile 0..3, l half 0..1
  v8f acc0 = {}, acc1 = {};
  int arow = tid >> 2, acs = (tid & 3) * 8;   // A loader mapping
  int bl   = tid & 63, bc0 = (tid >> 6) * 8;  // B loader mapping
  for (int kb = 0; kb < 24; ++kb) {
    int t = kb >> 3, cb = (kb & 7) * 32;
    // A tile (weights): 16B per thread, async global->LDS when available
    copy16_g2l(&A[arow * TS + acs],
               Wffb + (size_t)t * Cch * Cch + (size_t)(ob + arow) * Cch + cb + acs);
    { // B tile: 64 l x 32 c from x (f32 -> bf16), clamp+select padding (branch-free)
      long ls  = (long)lb + bl + (long)(t - 1) * 64;
      float zm = (ls >= 0 && ls < Lln) ? 1.0f : 0.0f;
      long lsc = ls < 0 ? 0 : (ls >= Lln ? (long)Lln - 1 : ls);
      __bf16 tmp[8];
      for (int i = 0; i < 8; ++i) {
        float v = x[(size_t)(cb + bc0 + i) * Lln + lsc] * zm;
        tmp[i] = (__bf16)v;
      }
      *reinterpret_cast<uint4*>(&B[bl * TS + bc0]) = *reinterpret_cast<const uint4*>(tmp);
      int kn = (kb + 1 < 24) ? kb + 1 : kb;
      __builtin_prefetch(Wffb + (size_t)(kn >> 3) * Cch * Cch +
                         (size_t)(ob + arow) * Cch + (kn & 7) * 32, 0, 3);
    }
    lds_copy_fence();
    __syncthreads();
    v16bf af  = frag_a(&A[(wo * 16 + n) * TS], half);
    v16bf bf0 = frag_b(&B[(wl * 32 + n) * TS], half);
    v16bf bf1 = frag_b(&B[(wl * 32 + 16 + n) * TS], half);
    acc0 = wmma_bf(af, bf0, acc0);
    acc1 = wmma_bf(af, bf1, acc1);
    __syncthreads();
  }
  for (int nt = 0; nt < 2; ++nt) {
    v8f& acc = nt ? acc1 : acc0;
    for (int r = 0; r < 8; ++r) {
      int oc = ob + wo * 16 + r + 8 * half;
      int l  = lb + (wl * 2 + nt) * 16 + n;
      float v = acc[r] + bff[oc];
      v = v > 0.f ? v : 0.f;
      outf[(size_t)oc * Lln + l] = v;
      float s = v, q = v * v;
      for (int m = 1; m < 16; m <<= 1) { s += __shfl_xor(s, m, 32); q += __shfl_xor(q, m, 32); }
      if (n == 0) {                  // unique deterministic slot per (oc, 16-l group)
        size_t slot = (size_t)oc * 8192 + (size_t)blockIdx.x * 4 + wl * 2 + nt;
        psum[slot] = s; psq[slot] = q;
      }
    }
  }
}

// ---- K2: reduce partials -> mu, rsig ---------------------------------------
__global__ __launch_bounds__(256) void k_stats(const float* __restrict__ psum,
    const float* __restrict__ psq, float* __restrict__ mu, float* __restrict__ rsig) {
  __shared__ float s1[256], s2[256];
  int c = blockIdx.x, t = threadIdx.x;
  float a = 0.f, b = 0.f;
  for (int i = t; i < 8192; i += 256) { a += psum[(size_t)c * 8192 + i]; b += psq[(size_t)c * 8192 + i]; }
  s1[t] = a; s2[t] = b; __syncthreads();
  for (int st = 128; st > 0; st >>= 1) {
    if (t < st) { s1[t] += s1[t + st]; s2[t] += s2[t + st]; }
    __syncthreads();
  }
  if (t == 0) {
    float m = s1[0] / (float)Lln;
    float v = s2[0] / (float)Lln - m * m;
    mu[c] = m; rsig[c] = rsqrtf(v + 1e-5f);
  }
}

// ---- K3: fold instance-norm into Wq/Wk (bf16) and bq/bk ---------------------
__global__ __launch_bounds__(256) void k_fold(const float* __restrict__ Wq,
    const float* __restrict__ bq, const float* __restrict__ Wk, const float* __restrict__ bk,
    const float* __restrict__ mu, const float* __restrict__ rsig,
    __bf16* __restrict__ Wqb, __bf16* __restrict__ Wkb,
    float* __restrict__ bqf, float* __restrict__ bkf) {
  int t = threadIdx.x;                 // waves 0-3 = q, waves 4-7 = k
  const float* W = (t < 128) ? Wq : Wk;
  const float* b = (t < 128) ? bq : bk;
  __bf16* Wb = (t < 128) ? Wqb : Wkb;
  float*  bo = (t < 128) ? bqf : bkf;
  int o = t & 127;
  float dot = 0.f;
  for (int c = 0; c < Cch; ++c) {
    float ws = W[o * Cch + c] * rsig[c];
    Wb[o * Cch + c] = (__bf16)ws;
    dot += ws * mu[c];
  }
  bo[o] = b[o] - dot;
}

// ---- K4: q/k/v projections (z = 0/1/2) -> bf16 [128][L] ---------------------
__global__ __launch_bounds__(256) void k_qkv(const float* __restrict__ outf,
    const float* __restrict__ f,
    const __bf16* __restrict__ Wqb, const __bf16* __restrict__ Wkb, const __bf16* __restrict__ Wvb,
    const float* __restrict__ bqf, const float* __restrict__ bkf, const float* __restrict__ bv,
    __bf16* __restrict__ qb, __bf16* __restrict__ kbuf, __bf16* __restrict__ vbuf) {
  __shared__ __bf16 A[64 * TS];
  __shared__ __bf16 B[64 * TS];
  int z = blockIdx.z;
  const __bf16* Wb  = (z == 0) ? Wqb : (z == 1) ? Wkb : Wvb;
  const float*  bia = (z == 0) ? bqf : (z == 1) ? bkf : bv;
  const float*  src = (z == 2) ? f : outf;
  __bf16* dst = (z == 0) ? qb : (z == 1) ? kbuf : vbuf;
  int tid = threadIdx.x;
  int lb  = blockIdx.x * 64;
  int ob  = blockIdx.y * 64;
  int wave = tid >> 5, lane = tid & 31, half = lane >> 4, n = lane & 15;
  int wo = wave & 3, wl = wave >> 2;
  int arow = tid >> 2, acs = (tid & 3) * 8;
  int bl   = tid & 63, bc0 = (tid >> 6) * 8;
  v8f acc0 = {}, acc1 = {};
  for (int kb = 0; kb < 8; ++kb) {
    int cb = kb * 32;
    copy16_g2l(&A[arow * TS + acs], Wb + (size_t)(ob + arow) * Cch + cb + acs);
    {
      __bf16 tmp[8];
      for (int i = 0; i < 8; ++i)
        tmp[i] = (__bf16)src[(size_t)(cb + bc0 + i) * Lln + lb + bl];
      *reinterpret_cast<uint4*>(&B[bl * TS + bc0]) = *reinterpret_cast<const uint4*>(tmp);
    }
    lds_copy_fence();
    __syncthreads();
    v16bf af  = frag_a(&A[(wo * 16 + n) * TS], half);
    v16bf bf0 = frag_b(&B[(wl * 32 + n) * TS], half);
    v16bf bf1 = frag_b(&B[(wl * 32 + 16 + n) * TS], half);
    acc0 = wmma_bf(af, bf0, acc0);
    acc1 = wmma_bf(af, bf1, acc1);
    __syncthreads();
  }
  for (int nt = 0; nt < 2; ++nt) {
    v8f& acc = nt ? acc1 : acc0;
    for (int r = 0; r < 8; ++r) {
      int oc = ob + wo * 16 + r + 8 * half;
      int l  = lb + (wl * 2 + nt) * 16 + n;
      dst[(size_t)oc * Lln + l] = (__bf16)(acc[r] + bia[oc]);
    }
  }
}

// ---- K5a: windowed attention weights (energy WMMA + in-register softmax) ----
__global__ __launch_bounds__(128) void k_att(const __bf16* __restrict__ qb,
    const __bf16* __restrict__ kb_, const float* __restrict__ mask,
    __bf16* __restrict__ attw) {
  __shared__ __bf16 QT[64 * TS2];    // [l][c]
  __shared__ __bf16 KT[128 * TS2];   // [m][c]
  __shared__ float  MV[128];
  int nb = blockIdx.x, tid = threadIdx.x;
  long base  = (long)nb * 64;
  long kbase = base - 32;
  for (int i = tid; i < 64 * 128; i += 128) {
    int l = i & 63, c = i >> 6;
    QT[l * TS2 + c] = qb[(size_t)c * Lln + base + l];
  }
  for (int i = tid; i < 128 * 128; i += 128) {
    int m = i & 127, c = i >> 7;
    long kp = kbase + m;
    float zm = (kp >= 0 && kp < Lln) ? 1.0f : 0.0f;
    long kpc = kp < 0 ? 0 : (kp >= Lln ? (long)Lln - 1 : kp);
    KT[m * TS2 + c] = (__bf16)((float)kb_[(size_t)c * Lln + kpc] * zm);
  }
  {
    long kp = kbase + tid;
    MV[tid] = (kp >= 0 && kp < Lln) ? mask[kp] : 0.f;
  }
  __syncthreads();
  int wave = tid >> 5, lane = tid & 31, half = lane >> 4, n = lane & 15;
  v8f z = {};
  v8f acc[8];
  for (int mt = 0; mt < 8; ++mt) acc[mt] = z;
  for (int kc = 0; kc < 4; ++kc) {           // K = channel, chunks of 32
    v16bf af = frag_a(&QT[(wave * 16 + n) * TS2 + kc * 32], half);
    for (int mt = 0; mt < 8; ++mt) {
      v16bf bf = frag_b(&KT[(mt * 16 + n) * TS2 + kc * 32], half);
      acc[mt] = wmma_bf(af, bf, acc[mt]);
    }
  }
  const float scale = 0.08838834764831845f;  // 1/sqrt(CQ)
  for (int r = 0; r < 8; ++r) {
    int l = wave * 16 + r + 8 * half;        // query row owned by this (r, half)
    float ev[8];
    float rmax = -1e30f;
    for (int mt = 0; mt < 8; ++mt) {
      int m = mt * 16 + n;
      int d = m - l;
      float fmv = (d >= 0 && d < 64) ? MV[m] : 0.f;
      float e = acc[mt][r] * scale + __logf(fmv + 1e-6f);
      ev[mt] = e;
      rmax = fmaxf(rmax, e);
    }
    for (int mk = 1; mk < 16; mk <<= 1) rmax = fmaxf(rmax, __shfl_xor(rmax, mk, 32));
    float rsum = 0.f;
    for (int mt = 0; mt < 8; ++mt) { ev[mt] = __expf(ev[mt] - rmax); rsum += ev[mt]; }
    for (int mk = 1; mk < 16; mk <<= 1) rsum += __shfl_xor(rsum, mk, 32);
    float inv = 1.0f / rsum;
    for (int mt = 0; mt < 8; ++mt) {
      int m = mt * 16 + n;
      int d = m - l;
      float fmv = (d >= 0 && d < 64) ? MV[m] : 0.f;
      attw[((size_t)nb * 64 + l) * 128 + m] = (__bf16)(ev[mt] * inv * fmv);
    }
  }
}

// ---- K5b: ov = v*att, relu, oo = Wo*rv + bo, h = oo*mask + out (bf16) -------
__global__ __launch_bounds__(256) void k_attout(const __bf16* __restrict__ vb,
    const __bf16* __restrict__ attw, const __bf16* __restrict__ Wob,
    const float* __restrict__ bo, const float* __restrict__ outf,
    const float* __restrict__ mask, __bf16* __restrict__ hb) {
  __shared__ __bf16 VA[128 * TS2];   // [c][m]
  __shared__ __bf16 RV[64 * TS2];    // [l][c] = relu(ov)^T in B layout
  int nb = blockIdx.x, tid = threadIdx.x;
  long base = (long)nb * 64, kbase = base - 32;
  if (nb > 0 && nb < (int)gridDim.x - 1) {
    // interior block: window fully in range, layout-preserving 16B copies
    for (int g = tid; g < 2048; g += 256) {
      int c = g >> 4, mg = (g & 15) * 8;
      copy16_g2l(&VA[c * TS2 + mg], vb + (size_t)c * Lln + kbase + mg);
    }
    lds_copy_fence();
  } else {
    for (int i = tid; i < 128 * 128; i += 256) {
      int m = i & 127, c = i >> 7;
      long kp = kbase + m;
      VA[c * TS2 + m] = (kp >= 0 && kp < Lln) ? vb[(size_t)c * Lln + kp] : (__bf16)0.f;
    }
  }
  __syncthreads();
  int wave = tid >> 5, lane = tid & 31, half = lane >> 4, n = lane & 15;
  // ov GEMM: M = c (8 tiles -> wave), N = l (4 tiles loop), K = m (4 chunks)
  for (int nt = 0; nt < 4; ++nt) {
    v8f acc = {};
    const __bf16* brow = attw + ((size_t)nb * 64 + nt * 16 + n) * 128;
    for (int kk = 0; kk < 4; ++kk) {
      v16bf af = frag_a(&VA[(wave * 16 + n) * TS2 + kk * 32], half);
      v16bf bf = frag_b(brow + kk * 32, half);
      acc = wmma_bf(af, bf, acc);
    }
    for (int r = 0; r < 8; ++r) {
      int c = wave * 16 + r + 8 * half;
      int l = nt * 16 + n;
      float v = acc[r];
      RV[l * TS2 + c] = (__bf16)(v > 0.f ? v : 0.f);
    }
  }
  __syncthreads();
  // oo GEMM: M = oc (16 tiles -> 2 per wave), N = l (4 tiles), K = c (4 chunks)
  for (int ot = 0; ot < 2; ++ot) {
    int octile = wave * 2 + ot;
    const __bf16* arow = Wob + (size_t)(octile * 16 + n) * 128;
    for (int nt = 0; nt < 4; ++nt) {
      v8f acc = {};
      const __bf16* brow = &RV[(nt * 16 + n) * TS2];
      for (int kk = 0; kk < 4; ++kk) {
        v16bf af = frag_a(arow + kk * 32, half);
        v16bf bf = frag_b(brow + kk * 32, half);
        acc = wmma_bf(af, bf, acc);
      }
      for (int r = 0; r < 8; ++r) {
        int oc = octile * 16 + r + 8 * half;
        long gl = base + nt * 16 + n;
        float v = (acc[r] + bo[oc]) * mask[gl] + outf[(size_t)oc * Lln + gl];
        hb[(size_t)oc * Lln + gl] = (__bf16)v;
      }
    }
  }
}

// ---- K6: y = (x + Wc*h + bc) * mask -----------------------------------------
__global__ __launch_bounds__(256) void k_final(const __bf16* __restrict__ hb,
    const __bf16* __restrict__ Wcb, const float* __restrict__ bc,
    const float* __restrict__ x, const float* __restrict__ mask, float* __restrict__ y) {
  __shared__ __bf16 A[64 * TS];
  __shared__ __bf16 B[64 * TS];
  int tid = threadIdx.x;
  int lb  = blockIdx.x * 64;
  int ob  = blockIdx.y * 64;
  int wave = tid >> 5, lane = tid & 31, half = lane >> 4, n = lane & 15;
  int wo = wave & 3, wl = wave >> 2;
  int arow = tid >> 2, acs = (tid & 3) * 8;
  int bl   = tid & 63, bc0 = (tid >> 6) * 8;
  v8f acc0 = {}, acc1 = {};
  for (int kb = 0; kb < 8; ++kb) {
    int cb = kb * 32;
    copy16_g2l(&A[arow * TS + acs], Wcb + (size_t)(ob + arow) * Cch + cb + acs);
    {
      __bf16 tmp[8];
      for (int i = 0; i < 8; ++i)
        tmp[i] = hb[(size_t)(cb + bc0 + i) * Lln + lb + bl];
      *reinterpret_cast<uint4*>(&B[bl * TS + bc0]) = *reinterpret_cast<const uint4*>(tmp);
    }
    lds_copy_fence();
    __syncthreads();
    v16bf af  = frag_a(&A[(wo * 16 + n) * TS], half);
    v16bf bf0 = frag_b(&B[(wl * 32 + n) * TS], half);
    v16bf bf1 = frag_b(&B[(wl * 32 + 16 + n) * TS], half);
    acc0 = wmma_bf(af, bf0, acc0);
    acc1 = wmma_bf(af, bf1, acc1);
    __syncthreads();
  }
  for (int nt = 0; nt < 2; ++nt) {
    v8f& acc = nt ? acc1 : acc0;
    for (int r = 0; r < 8; ++r) {
      int oc = ob + wo * 16 + r + 8 * half;
      int l  = lb + (wl * 2 + nt) * 16 + n;
      size_t idx = (size_t)oc * Lln + l;
      y[idx] = (x[idx] + acc[r] + bc[oc]) * mask[l];
    }
  }
}

// ---- host launch ------------------------------------------------------------
extern "C" void kernel_launch(void* const* d_in, const int* in_sizes, int n_in,
                              void* d_out, int out_size, void* d_ws, size_t ws_size,
                              hipStream_t stream) {
  const float* x    = (const float*)d_in[0];
  const float* f    = (const float*)d_in[1];
  const float* mask = (const float*)d_in[2];
  const float* Wff  = (const float*)d_in[3];
  const float* bff  = (const float*)d_in[4];
  const float* Wq   = (const float*)d_in[5];
  const float* bq   = (const float*)d_in[6];
  const float* Wk   = (const float*)d_in[7];
  const float* bk   = (const float*)d_in[8];
  const float* Wv   = (const float*)d_in[9];
  const float* bv   = (const float*)d_in[10];
  const float* Wo   = (const float*)d_in[11];
  const float* bo   = (const float*)d_in[12];
  const float* Wc   = (const float*)d_in[13];
  const float* bc   = (const float*)d_in[14];
  float* y = (float*)d_out;

  char* ws = (char*)d_ws;
  size_t off = 0;
  auto alloc = [&](size_t bytes) -> char* {
    char* p = ws + off;
    off += (bytes + 255) & ~(size_t)255;
    return p;
  };
  float*  outf = (float*) alloc((size_t)Cch * Lln * 4);   // 134 MB
  __bf16* qb   = (__bf16*)alloc((size_t)128 * Lln * 2);   // 33.5 MB
  __bf16* kbuf = (__bf16*)alloc((size_t)128 * Lln * 2);
  __bf16* vbuf = (__bf16*)alloc((size_t)128 * Lln * 2);
  __bf16* hb   = (__bf16*)alloc((size_t)Cch * Lln * 2);   // 67 MB
  __bf16* attw = (__bf16*)alloc((size_t)2048 * 64 * 128 * 2); // 33.5 MB
  float*  psum = (float*) alloc((size_t)Cch * 8192 * 4);  // 8 MB
  float*  psq  = (float*) alloc((size_t)Cch * 8192 * 4);
  __bf16* Wffb = (__bf16*)alloc((size_t)3 * Cch * Cch * 2);
  __bf16* Wvb  = (__bf16*)alloc((size_t)128 * Cch * 2);
  __bf16* Wob  = (__bf16*)alloc((size_t)Cch * 128 * 2);
  __bf16* Wcb  = (__bf16*)alloc((size_t)Cch * Cch * 2);
  __bf16* Wqb  = (__bf16*)alloc((size_t)128 * Cch * 2);
  __bf16* Wkb  = (__bf16*)alloc((size_t)128 * Cch * 2);
  float*  bqf  = (float*) alloc(128 * 4);
  float*  bkf  = (float*) alloc(128 * 4);
  float*  mu   = (float*) alloc(Cch * 4);
  float*  rsig = (float*) alloc(Cch * 4);

  k_prep <<<1280, 256, 0, stream>>>(Wff, Wv, Wo, Wc, Wffb, Wvb, Wob, Wcb);
  k_conv <<<dim3(Lln / 64, Cch / 64), 256, 0, stream>>>(x, Wffb, bff, outf, psum, psq);
  k_stats<<<Cch, 256, 0, stream>>>(psum, psq, mu, rsig);
  k_fold <<<1, 256, 0, stream>>>(Wq, bq, Wk, bk, mu, rsig, Wqb, Wkb, bqf, bkf);
  k_qkv  <<<dim3(Lln / 64, 2, 3), 256, 0, stream>>>(outf, f, Wqb, Wkb, Wvb, bqf, bkf, bv,
                                                    qb, kbuf, vbuf);
  k_att  <<<2048, 128, 0, stream>>>(qb, kbuf, mask, attw);
  k_attout<<<2048, 256, 0, stream>>>(vbuf, attw, Wob, bo, outf, mask, hb);
  k_final<<<dim3(Lln / 64, Cch / 64), 256, 0, stream>>>(hb, Wcb, bc, x, mask, y);
}
